// DsaFlashMla_2911987826915
// MI455X (gfx1250) — 16-bit and 8-bit matrix operand order repaired and checked by reading, compiled, not run on hardware
//
#include <hip/hip_runtime.h>
#include <cmath>
#include <stdint.h>

typedef __attribute__((ext_vector_type(8)))  float  v8f;
typedef __attribute__((ext_vector_type(4)))  float  f4;
typedef __attribute__((ext_vector_type(16))) int    v16i;
typedef __attribute__((ext_vector_type(16))) __bf16 v16bf;

union BF16x16 { v16bf v; unsigned short u[16]; };
union I32x16  { v16i  v; unsigned       u[16]; };

#define SOFTMAX_SCALE_F 0.13523391f
#define INV_SQRT_ID     0.08838834764831845f
#define NEG_BIG         -3.0e38f

__device__ __forceinline__ unsigned short bf16_bits(float f) {
  unsigned u = __float_as_uint(f);
  unsigned r = (u + 0x7FFFu + ((u >> 16) & 1u)) >> 16;
  return (unsigned short)r;
}

__device__ __forceinline__ unsigned char e4m3_bits(float x) {
  float a = fabsf(x);
  unsigned s = (__float_as_uint(x) >> 24) & 0x80u;
  if (!(a == a)) return (unsigned char)(s | 0x7Fu);
  if (a >= 448.f) return (unsigned char)(s | 0x7Eu);
  int e = (int)((__float_as_uint(a) >> 23) & 0xFF) - 127;
  unsigned bits;
  if (a == 0.f) return (unsigned char)s;
  if (e < -6) {
    bits = (unsigned)rintf(ldexpf(a, 9));
  } else {
    float mant = ldexpf(a, -e);
    unsigned m3 = (unsigned)rintf((mant - 1.f) * 8.f);
    bits = ((unsigned)(e + 7) << 3) + m3;
  }
  if (bits > 0x7Eu) bits = 0x7Eu;
  return (unsigned char)(s | bits);
}

__device__ __forceinline__ f4 ldg4(const float* p) { return *(const f4*)p; }

__device__ __forceinline__ unsigned pack4_e4m3(f4 w) {
#if __has_builtin(__builtin_amdgcn_cvt_pk_fp8_f32)
  int d = 0;
  d = __builtin_amdgcn_cvt_pk_fp8_f32(w.x, w.y, d, false);
  d = __builtin_amdgcn_cvt_pk_fp8_f32(w.z, w.w, d, true);
  return (unsigned)d;
#else
  return (unsigned)e4m3_bits(w.x) | ((unsigned)e4m3_bits(w.y) << 8) |
         ((unsigned)e4m3_bits(w.z) << 16) | ((unsigned)e4m3_bits(w.w) << 24);
#endif
}

__device__ __forceinline__ void put8_bf16(BF16x16& V, int ebase, const float* p) {
  f4 a = ldg4(p), b = ldg4(p + 4);
  V.u[ebase + 0] = bf16_bits(a.x); V.u[ebase + 1] = bf16_bits(a.y);
  V.u[ebase + 2] = bf16_bits(a.z); V.u[ebase + 3] = bf16_bits(a.w);
  V.u[ebase + 4] = bf16_bits(b.x); V.u[ebase + 5] = bf16_bits(b.y);
  V.u[ebase + 6] = bf16_bits(b.z); V.u[ebase + 7] = bf16_bits(b.w);
}
__device__ __forceinline__ void put8_bf16s(BF16x16& V, int ebase, const float* p, float s) {
  f4 a = ldg4(p), b = ldg4(p + 4);
  V.u[ebase + 0] = bf16_bits(a.x * s); V.u[ebase + 1] = bf16_bits(a.y * s);
  V.u[ebase + 2] = bf16_bits(a.z * s); V.u[ebase + 3] = bf16_bits(a.w * s);
  V.u[ebase + 4] = bf16_bits(b.x * s); V.u[ebase + 5] = bf16_bits(b.y * s);
  V.u[ebase + 6] = bf16_bits(b.z * s); V.u[ebase + 7] = bf16_bits(b.w * s);
}
__device__ __forceinline__ void put8_zero(BF16x16& V, int ebase) {
  #pragma unroll
  for (int i = 0; i < 8; ++i) V.u[ebase + i] = 0;
}

__device__ __forceinline__ void rope_cs(int i, int pos, float* c, float* s) {
  float inv = powf(10000.f, -(float)i / 32.f);
  float ang = (float)pos * inv;
  sincosf(ang, s, c);
}

__device__ __forceinline__ v8f vzero8() {
  v8f z;
  #pragma unroll
  for (int i = 0; i < 8; ++i) z[i] = 0.f;
  return z;
}

__device__ __forceinline__ v16i load_a_fp8(const unsigned char* A, int ldK, int kbase, int lane) {
  int row = lane & 15, hi = lane >> 4;
  I32x16 a;
  #pragma unroll
  for (int d = 0; d < 16; ++d) {
    int c = d >> 3, v = d & 7;
    int k = kbase + ((d >> 2) << 5) + (hi << 4) + ((d & 3) << 2);
    a.u[d] = *(const unsigned*)(A + (size_t)row * ldK + k);
  }
  return a.v;
}

__device__ __forceinline__ v16i load_b_fp8_f32row(const float* wrow, int kbase, int lane) {
  int hi = lane >> 4;
  I32x16 b;
  #pragma unroll
  for (int v = 0; v < 16; ++v) {
    int k = kbase + ((v >> 2) << 5) + (hi << 4) + ((v & 3) << 2);
    b.u[v] = pack4_e4m3(ldg4(wrow + k));
  }
  return b.v;
}

__global__ void k_zero32(unsigned* p, int n) {
  int i = blockIdx.x * blockDim.x + threadIdx.x;
  if (i < n) p[i] = 0u;
}

__global__ void k_quant_rows(const float* X, unsigned char* Q, float* SC, int K) {
  int b = blockIdx.x, kb = blockIdx.y, t = threadIdx.x;
  __shared__ float red[128];
  float x = X[(size_t)b * K + kb * 128 + t];
  red[t] = fabsf(x);
  __syncthreads();
  for (int s = 64; s > 0; s >>= 1) { if (t < s) red[t] = fmaxf(red[t], red[t + s]); __syncthreads(); }
  float mx = fmaxf(red[0], 1e-4f);
  float sc = exp2f(ceilf(log2f(mx / 448.f)));
  if (t == 0) SC[b * (K >> 7) + kb] = sc;
  Q[(size_t)b * K + kb * 128 + t] = e4m3_bits(x / sc);
}

__global__ void k_gemm_fp8(const unsigned char* Aq, const float* Asc,
                           const float* W, const float* Wsc,
                           float* Y, int N, int K, int ldY) {
  int lane = threadIdx.x;
  int nt = blockIdx.x;
  int nl = lane & 15, hi = lane >> 4;
  int n = nt * 16 + nl;
  int KB = K >> 7;
  int nblk = nt >> 3;
  float acc[8];
  #pragma unroll
  for (int r = 0; r < 8; ++r) acc[r] = 0.f;
  const float* wrow = W + (size_t)n * K;
  for (int kb = 0; kb < KB; ++kb) {
    if (kb + 1 < KB) __builtin_prefetch(wrow + ((kb + 1) << 7), 0, 1);
    v16i a = load_a_fp8(Aq, K, kb << 7, lane);
    v16i b = load_b_fp8_f32row(wrow, kb << 7, lane);
    v8f c = vzero8();
    c = __builtin_amdgcn_wmma_f32_16x16x128_fp8_fp8(a, b, (short)0, c, false, false);
    float wsc = Wsc[nblk * KB + kb];
    #pragma unroll
    for (int r = 0; r < 8; ++r) {
      int m = r + (hi << 3);
      float asc = (m < 4) ? Asc[m * KB + kb] : 0.f;
      acc[r] += c[r] * asc * wsc;
    }
  }
  #pragma unroll
  for (int r = 0; r < 8; ++r) {
    int m = r + (hi << 3);
    if (m < 4) Y[(size_t)m * ldY + n] = acc[r];
  }
}

__global__ void k_rmsnorm(float* X, const float* w, int K) {
  int b = blockIdx.x, t = threadIdx.x;
  __shared__ float red[256];
  float ss = 0.f;
  for (int k = t; k < K; k += 256) { float v = X[(size_t)b * K + k]; ss += v * v; }
  red[t] = ss; __syncthreads();
  for (int s = 128; s > 0; s >>= 1) { if (t < s) red[t] += red[t + s]; __syncthreads(); }
  float r = rsqrtf(red[0] / (float)K + 1e-6f);
  for (int k = t; k < K; k += 256) X[(size_t)b * K + k] = X[(size_t)b * K + k] * r * w[k];
}

__global__ void k_gemv4(const float* X, const float* W, float* Y, int N, int K, float scale) {
  int n = blockIdx.x, t = threadIdx.x;
  float a0 = 0.f, a1 = 0.f, a2 = 0.f, a3 = 0.f;
  const float* wr = W + (size_t)n * K;
  for (int k = t * 4; k < K; k += 1024) {
    f4 w = ldg4(wr + k);
    f4 x0 = ldg4(X + k), x1 = ldg4(X + K + k), x2 = ldg4(X + 2 * K + k), x3 = ldg4(X + 3 * K + k);
    a0 += w.x * x0.x + w.y * x0.y + w.z * x0.z + w.w * x0.w;
    a1 += w.x * x1.x + w.y * x1.y + w.z * x1.z + w.w * x1.w;
    a2 += w.x * x2.x + w.y * x2.y + w.z * x2.z + w.w * x2.w;
    a3 += w.x * x3.x + w.y * x3.y + w.z * x3.z + w.w * x3.w;
  }
  __shared__ float red[4][256];
  red[0][t] = a0; red[1][t] = a1; red[2][t] = a2; red[3][t] = a3;
  __syncthreads();
  for (int s = 128; s > 0; s >>= 1) {
    if (t < s) { for (int b = 0; b < 4; ++b) red[b][t] += red[b][t + s]; }
    __syncthreads();
  }
  if (t < 4) Y[(size_t)t * N + n] = red[t][0] * scale;
}

__global__ void k_rope_q(float* q_raw, const int* ctx) {
  int b = blockIdx.x, h = blockIdx.y, i = threadIdx.x;
  if (i < 32) {
    int pos = ctx[0]; float c, s; rope_cs(i, pos, &c, &s);
    float* qp = q_raw + ((size_t)b * 128 + h) * 192 + 128;
    float xr = qp[2 * i], xi = qp[2 * i + 1];
    qp[2 * i] = xr * c - xi * s;
    qp[2 * i + 1] = xr * s + xi * c;
  }
}

__global__ void k_kv_post(float* kv, const float* kvnw, const int* ctx) {
  int b = blockIdx.x, t = threadIdx.x;
  __shared__ float red[256];
  float ss = 0.f;
  for (int k = t; k < 512; k += 256) { float v = kv[b * 576 + k]; ss += v * v; }
  red[t] = ss; __syncthreads();
  for (int s = 128; s > 0; s >>= 1) { if (t < s) red[t] += red[t + s]; __syncthreads(); }
  float r = rsqrtf(red[0] / 512.f + 1e-6f);
  for (int k = t; k < 512; k += 256) kv[b * 576 + k] *= r * kvnw[k];
  if (t < 32) {
    int pos = ctx[0]; float c, s; rope_cs(t, pos, &c, &s);
    float xr = kv[b * 576 + 512 + 2 * t], xi = kv[b * 576 + 512 + 2 * t + 1];
    kv[b * 576 + 512 + 2 * t] = xr * c - xi * s;
    kv[b * 576 + 512 + 2 * t + 1] = xr * s + xi * c;
  }
}

__device__ __forceinline__ void rope_half_hadamard_128(float* sh, int t, int pos) {
  float nv;
  if (t < 32)      { float c, s; rope_cs(t, pos, &c, &s);       nv = sh[t] * c - sh[t + 32] * s; }
  else if (t < 64) { int d = t - 32; float c, s; rope_cs(d, pos, &c, &s); nv = sh[d] * s + sh[t] * c; }
  else             nv = sh[t];
  __syncthreads(); sh[t] = nv; __syncthreads();
  for (int h = 1; h < 128; h <<= 1) {
    int j = t & (2 * h - 1); int base = t - j; int jj = (j < h) ? j : (j - h);
    float a = sh[base + jj], b = sh[base + jj + h];
    float v = (j < h) ? (a + b) : (a - b);
    __syncthreads(); sh[t] = v; __syncthreads();
  }
}

__global__ void k_ki_post(float* ki, const float* w, const float* bb, const int* ctx) {
  __shared__ float sh[128]; __shared__ float red[128];
  int b = blockIdx.x, t = threadIdx.x;
  float x = ki[b * 128 + t];
  red[t] = x; __syncthreads();
  for (int s = 64; s > 0; s >>= 1) { if (t < s) red[t] += red[t + s]; __syncthreads(); }
  float mean = red[0] / 128.f; __syncthreads();
  float d = x - mean; red[t] = d * d; __syncthreads();
  for (int s = 64; s > 0; s >>= 1) { if (t < s) red[t] += red[t + s]; __syncthreads(); }
  float var = red[0] / 128.f;
  sh[t] = d * rsqrtf(var + 1e-6f) * w[t] + bb[t];
  __syncthreads();
  rope_half_hadamard_128(sh, t, ctx[0]);
  ki[b * 128 + t] = sh[t] * INV_SQRT_ID;
}

__global__ void k_qi_post(float* qi, const int* ctx) {
  __shared__ float sh[128];
  int b = blockIdx.x, h = blockIdx.y, t = threadIdx.x;
  float* v = qi + ((size_t)b * 64 + h) * 128;
  sh[t] = v[t]; __syncthreads();
  rope_half_hadamard_128(sh, t, ctx[0]);
  v[t] = sh[t] * INV_SQRT_ID;
}

__global__ void k_score(const float* qi, const float* ki_new, const float* w_head,
                        const float* idx_cache, const int* btab, const int* ctx, float* score) {
  int lane = threadIdx.x;
  int tile = blockIdx.x, b = blockIdx.y;
  int pos = ctx[0], T = pos + 1;
  int nl = lane & 15, hi = lane >> 4;
  int t = tile * 16 + nl;
  const float* krow = nullptr;
  if (t < T) krow = (t == pos) ? (ki_new + b * 128)
                               : (idx_cache + (size_t)(btab[b * 65 + (t >> 6)] * 64 + (t & 63)) * 128);
  float part = 0.f;
  for (int mt = 0; mt < 4; ++mt) {
    const float* qrow = qi + ((size_t)b * 64 + mt * 16 + nl) * 128;
    v8f acc = vzero8();
    for (int kk = 0; kk < 4; ++kk) {
      BF16x16 A, Bm;
      put8_bf16(A, 0, qrow + kk * 32 + (hi << 4));
      put8_bf16(A, 8, qrow + kk * 32 + (hi << 4) + 8);
      if (krow) {
        put8_bf16(Bm, 0, krow + kk * 32 + (hi << 4));
        put8_bf16(Bm, 8, krow + kk * 32 + (hi << 4) + 8);
      } else { put8_zero(Bm, 0); put8_zero(Bm, 8); }
      acc = __builtin_amdgcn_wmma_f32_16x16x32_bf16(false, A.v, false, Bm.v, (short)0, acc, false, false);
    }
    #pragma unroll
    for (int r = 0; r < 8; ++r) {
      int h = mt * 16 + r + (hi << 3);
      part += w_head[b * 64 + h] * fmaxf(acc[r] * INV_SQRT_ID, 0.f);
    }
  }
  part += __shfl_xor(part, 16, 32);
  if (hi == 0) score[b * 4160 + t] = (t < T) ? part : NEG_BIG;
}

__global__ void k_topk(const float* score, int* top, const int* ctx) {
  __shared__ unsigned hist[256];
  __shared__ unsigned sh_prefix; __shared__ int sh_k; __shared__ int sh_cnt; __shared__ int sh_cnt2;
  int b = blockIdx.x, t = threadIdx.x;
  int T = ctx[0] + 1;
  const float* sc = score + b * 4160;
  if (t == 0) { sh_prefix = 0u; sh_k = 2048; }
  __syncthreads();
  for (int round = 3; round >= 0; --round) {
    hist[t] = 0u; __syncthreads();
    unsigned mask = (round == 3) ? 0u : (0xFFFFFFFFu << ((round + 1) * 8));
    for (int i = t; i < 4160; i += 256) {
      if (i >= T) continue;
      unsigned u = __float_as_uint(sc[i]);
      unsigned key = (u & 0x80000000u) ? ~u : (u | 0x80000000u);
      if ((key & mask) == (sh_prefix & mask)) atomicAdd(&hist[(key >> (round * 8)) & 0xFF], 1u);
    }
    __syncthreads();
    if (t == 0) {
      int k = sh_k;
      for (int bin = 255; bin >= 0; --bin) {
        int c = (int)hist[bin];
        if (k <= c) { sh_prefix |= ((unsigned)bin) << (round * 8); sh_k = k; break; }
        k -= c;
      }
    }
    __syncthreads();
  }
  if (t == 0) { sh_cnt = 0; sh_cnt2 = 0; }
  __syncthreads();
  unsigned thr = sh_prefix;
  for (int i = t; i < 4160; i += 256) {
    if (i >= T) continue;
    unsigned u = __float_as_uint(sc[i]);
    unsigned key = (u & 0x80000000u) ? ~u : (u | 0x80000000u);
    if (key > thr) { int p = atomicAdd(&sh_cnt, 1); if (p < 2048) top[b * 2048 + p] = i; }
  }
  __syncthreads();
  int base = sh_cnt;
  for (int i = t; i < 4160; i += 256) {
    if (i >= T) continue;
    unsigned u = __float_as_uint(sc[i]);
    unsigned key = (u & 0x80000000u) ? ~u : (u | 0x80000000u);
    if (key == thr) { int p = atomicAdd(&sh_cnt2, 1); if (base + p < 2048) top[b * 2048 + base + p] = i; }
  }
}

__global__ void k_qabs(const float* q_raw, const float* wkvb, const float* wkvbs, float* qabs) {
  int lane = threadIdx.x;
  int h = blockIdx.x, nt = blockIdx.y;
  int nl = lane & 15, hi = lane >> 4;
  int n = nt * 16 + nl;
  float s = wkvbs[(2 * h) * 4 + (n >> 7)];
  v8f acc = vzero8();
  for (int kk = 0; kk < 4; ++kk) {
    BF16x16 A, Bm;
    if (nl < 4) {
      const float* ar = q_raw + ((size_t)nl * 128 + h) * 192 + kk * 32;
      put8_bf16(A, 0, ar + (hi << 4));
      put8_bf16(A, 8, ar + (hi << 4) + 8);
    } else { put8_zero(A, 0); put8_zero(A, 8); }
    #pragma unroll
    for (int e = 0; e < 16; ++e) {
      int kb = kk * 32 + (hi << 4) + e;
      Bm.u[e] = bf16_bits(wkvb[((size_t)h * 256 + kb) * 512 + n] * s);
    }
    acc = __builtin_amdgcn_wmma_f32_16x16x32_bf16(false, A.v, false, Bm.v, (short)0, acc, false, false);
  }
  if (hi == 0) {
    #pragma unroll
    for (int r = 0; r < 8; ++r) if (r < 4) qabs[((size_t)r * 128 + h) * 512 + n] = acc[r];
  }
}

#define LDC 584
__global__ void k_logits(const float* qabs, const float* q_raw, const float* kvcache,
                         const float* kv_new, const int* top, const int* btab, const int* ctx,
                         float* logits) {
  __shared__ unsigned short cs[16 * LDC];
  int jt = blockIdx.x, b = blockIdx.y;
  int tid = threadIdx.x;
  int pos = ctx[0];
  for (int idx = tid; idx < 16 * 144; idx += 256) {
    int row = idx / 144, c4 = (idx - row * 144) * 4;
    int tok = top[b * 2048 + jt * 16 + row];
    const float* kr = (tok == pos) ? (kv_new + b * 576)
        : (kvcache + (size_t)(btab[b * 65 + (tok >> 6)] * 64 + (tok & 63)) * 576);
    f4 v = ldg4(kr + c4);
    cs[row * LDC + c4 + 0] = bf16_bits(v.x);
    cs[row * LDC + c4 + 1] = bf16_bits(v.y);
    cs[row * LDC + c4 + 2] = bf16_bits(v.z);
    cs[row * LDC + c4 + 3] = bf16_bits(v.w);
  }
  __syncthreads();
  int w = tid >> 5, lane = tid & 31;
  int nl = lane & 15, hi = lane >> 4;
  int h0 = w * 16;
  const float* qa = qabs + ((size_t)b * 128 + h0 + nl) * 512;
  const float* qp = q_raw + ((size_t)b * 128 + h0 + nl) * 192 + 128;
  v8f acc = vzero8();
  for (int kk = 0; kk < 18; ++kk) {
    BF16x16 A, Bm;
    #pragma unroll
    for (int g = 0; g < 2; ++g) {
      int ka = kk * 32 + (hi << 4) + g * 8;
      const float* src = (ka < 512) ? (qa + ka) : (qp + (ka - 512));
      put8_bf16(A, g * 8, src);
    }
    #pragma unroll
    for (int e = 0; e < 16; ++e) {
      int kb = kk * 32 + (hi << 4) + e;
      Bm.u[e] = cs[nl * LDC + kb];
    }
    acc = __builtin_amdgcn_wmma_f32_16x16x32_bf16(false, A.v, false, Bm.v, (short)0, acc, false, false);
  }
  #pragma unroll
  for (int r = 0; r < 8; ++r) {
    int h = h0 + r + (hi << 3);
    logits[((size_t)b * 128 + h) * 2048 + jt * 16 + nl] = acc[r] * SOFTMAX_SCALE_F;
  }
}

__global__ void k_softmax(float* logits) {
  size_t row = (size_t)blockIdx.y * 128 + blockIdx.x;
  float* p = logits + row * 2048;
  __shared__ float red[256];
  int t = threadIdx.x;
  float m = NEG_BIG;
  for (int j = t; j < 2048; j += 256) m = fmaxf(m, p[j]);
  red[t] = m; __syncthreads();
  for (int s = 128; s > 0; s >>= 1) { if (t < s) red[t] = fmaxf(red[t], red[t + s]); __syncthreads(); }
  m = red[0]; __syncthreads();
  float sum = 0.f;
  for (int j = t; j < 2048; j += 256) { float e = __expf(p[j] - m); p[j] = e; sum += e; }
  red[t] = sum; __syncthreads();
  for (int s = 128; s > 0; s >>= 1) { if (t < s) red[t] += red[t + s]; __syncthreads(); }
  float inv = 1.f / red[0];
  for (int j = t; j < 2048; j += 256) p[j] *= inv;
}

__global__ void k_oc(const float* p, const float* kvcache, const float* kv_new,
                     const int* top, const int* btab, const int* ctx, float* ohc) {
  __shared__ unsigned short cg[32 * 16];
  int ct = blockIdx.x, b = blockIdx.y, tid = threadIdx.x;
  int pos = ctx[0];
  int w = tid >> 5, lane = tid & 31, nl = lane & 15, hi = lane >> 4;
  int h0 = w * 16;
  const float* prow = p + ((size_t)b * 128 + h0 + nl) * 2048;
  v8f acc = vzero8();
  for (int k0 = 0; k0 < 2048; k0 += 32) {
    if (tid < 128) {
      int tl = tid >> 2, c4 = (tid & 3) * 4;
      int tok = top[b * 2048 + k0 + tl];
      const float* kr = (tok == pos) ? (kv_new + b * 576)
          : (kvcache + (size_t)(btab[b * 65 + (tok >> 6)] * 64 + (tok & 63)) * 576);
      f4 v = ldg4(kr + ct * 16 + c4);
      cg[tl * 16 + c4 + 0] = bf16_bits(v.x);
      cg[tl * 16 + c4 + 1] = bf16_bits(v.y);
      cg[tl * 16 + c4 + 2] = bf16_bits(v.z);
      cg[tl * 16 + c4 + 3] = bf16_bits(v.w);
    }
    __syncthreads();
    BF16x16 A, Bm;
    put8_bf16(A, 0, prow + k0 + (hi << 4));
    put8_bf16(A, 8, prow + k0 + (hi << 4) + 8);
    #pragma unroll
    for (int e = 0; e < 16; ++e) {
      int kb = (hi << 4) + e;
      Bm.u[e] = cg[kb * 16 + nl];
    }
    acc = __builtin_amdgcn_wmma_f32_16x16x32_bf16(false, A.v, false, Bm.v, (short)0, acc, false, false);
    __syncthreads();
  }
  #pragma unroll
  for (int r = 0; r < 8; ++r) {
    int h = h0 + r + (hi << 3);
    ohc[((size_t)b * 128 + h) * 512 + ct * 16 + nl] = acc[r];
  }
}

__global__ void k_ov(const float* ohc, const float* wkvb, const float* wkvbs, float* o2) {
  int lane = threadIdx.x;
  int h = blockIdx.x, vt = blockIdx.y;
  int nl = lane & 15, hi = lane >> 4;
  int n = vt * 16 + nl;
  const float* wrow = wkvb + ((size_t)h * 256 + 128 + n) * 512;
  v8f acc = vzero8();
  for (int kk = 0; kk < 16; ++kk) {
    float s = wkvbs[(2 * h + 1) * 4 + (kk >> 2)];
    BF16x16 A, Bm;
    if (nl < 4) {
      const float* ar = ohc + ((size_t)nl * 128 + h) * 512 + kk * 32;
      put8_bf16(A, 0, ar + (hi << 4));
      put8_bf16(A, 8, ar + (hi << 4) + 8);
    } else { put8_zero(A, 0); put8_zero(A, 8); }
    put8_bf16s(Bm, 0, wrow + kk * 32 + (hi << 4), s);
    put8_bf16s(Bm, 8, wrow + kk * 32 + (hi << 4) + 8, s);
    acc = __builtin_amdgcn_wmma_f32_16x16x32_bf16(false, A.v, false, Bm.v, (short)0, acc, false, false);
  }
  if (hi == 0) {
    #pragma unroll
    for (int r = 0; r < 8; ++r) if (r < 4) o2[((size_t)r * 128 + h) * 128 + n] = acc[r];
  }
}

extern "C" void kernel_launch(void* const* d_in, const int* in_sizes, int n_in,
                              void* d_out, int out_size, void* d_ws, size_t ws_size,
                              hipStream_t stream) {
  (void)in_sizes; (void)n_in; (void)out_size; (void)ws_size;
  const float* x         = (const float*)d_in[0];
  const float* kv_cache  = (const float*)d_in[1];
  const float* idx_cache = (const float*)d_in[2];
  const int*   btab      = (const int*)d_in[3];
  const int*   ctx       = (const int*)d_in[4];
  const float* wq_a_w    = (const float*)d_in[5];
  const float* wq_a_s    = (const float*)d_in[6];
  const float* q_norm_w  = (const float*)d_in[7];
  const float* wq_b_w    = (const float*)d_in[8];
  const float* wq_b_s    = (const float*)d_in[9];
  const float* wkv_a_w   = (const float*)d_in[10];
  const float* kv_norm_w = (const float*)d_in[11];
  const float* wkv_b_w   = (const float*)d_in[12];
  const float* wkv_b_s   = (const float*)d_in[13];
  const float* idx_wq_w  = (const float*)d_in[14];
  const float* idx_wk_w  = (const float*)d_in[15];
  const float* idx_kn_w  = (const float*)d_in[16];
  const float* idx_kn_b  = (const float*)d_in[17];
  const float* idx_wp_w  = (const float*)d_in[18];
  const float* wo_w      = (const float*)d_in[19];
  const float* wo_s      = (const float*)d_in[20];

  char* w8 = (char*)d_ws;
  unsigned char* x_f8   = (unsigned char*)(w8 + 0);
  unsigned char* ql_f8  = (unsigned char*)(w8 + 114688);
  unsigned char* o_f8   = (unsigned char*)(w8 + 139264);
  float* x_sc   = (float*)(w8 + 401408);
  float* ql_sc  = (float*)(w8 + 402304);
  float* o_sc   = (float*)(w8 + 402496);
  float* q_lora = (float*)(w8 + 404544);
  float* q_raw  = (float*)(w8 + 429120);
  float* kv_new = (float*)(w8 + 822336);
  float* ki     = (float*)(w8 + 831552);
  float* w_head = (float*)(w8 + 833600);
  float* qi     = (float*)(w8 + 834624);
  float* score  = (float*)(w8 + 965696);
  int*   top    = (int*)  (w8 + 1032256);
  float* qabs   = (float*)(w8 + 1065024);
  float* logits = (float*)(w8 + 2113600);
  float* ohc    = (float*)(w8 + 6307904);
  float* o2     = (float*)(w8 + 7356480);

  k_zero32<<<392, 256, 0, stream>>>((unsigned*)d_ws, 100352);
  k_quant_rows<<<dim3(4, 56), 128, 0, stream>>>(x, x_f8, x_sc, 7168);
  k_gemm_fp8<<<96, 32, 0, stream>>>(x_f8, x_sc, wq_a_w, wq_a_s, q_lora, 1536, 7168, 1536);
  k_rmsnorm<<<4, 256, 0, stream>>>(q_lora, q_norm_w, 1536);
  k_quant_rows<<<dim3(4, 12), 128, 0, stream>>>(q_lora, ql_f8, ql_sc, 1536);
  k_gemm_fp8<<<1536, 32, 0, stream>>>(ql_f8, ql_sc, wq_b_w, wq_b_s, q_raw, 24576, 1536, 24576);
  k_rope_q<<<dim3(4, 128), 32, 0, stream>>>(q_raw, ctx);
  k_gemv4<<<576, 256, 0, stream>>>(x, wkv_a_w, kv_new, 576, 7168, 1.f);
  k_kv_post<<<4, 256, 0, stream>>>(kv_new, kv_norm_w, ctx);
  k_gemv4<<<128, 256, 0, stream>>>(x, idx_wk_w, ki, 128, 7168, 1.f);
  k_ki_post<<<4, 128, 0, stream>>>(ki, idx_kn_w, idx_kn_b, ctx);
  k_gemv4<<<64, 256, 0, stream>>>(x, idx_wp_w, w_head, 64, 7168, 0.125f);
  k_gemv4<<<8192, 256, 0, stream>>>(q_lora, idx_wq_w, qi, 8192, 1536, 1.f);
  k_qi_post<<<dim3(4, 64), 128, 0, stream>>>(qi, ctx);
  k_score<<<dim3(260, 4), 32, 0, stream>>>(qi, ki, w_head, idx_cache, btab, ctx, score);
  k_topk<<<4, 256, 0, stream>>>(score, top, ctx);
  k_qabs<<<dim3(128, 32), 32, 0, stream>>>(q_raw, wkv_b_w, wkv_b_s, qabs);
  k_logits<<<dim3(128, 4), 256, 0, stream>>>(qabs, q_raw, kv_cache, kv_new, top, btab, ctx, logits);
  k_softmax<<<dim3(128, 4), 256, 0, stream>>>(logits);
  k_oc<<<dim3(32, 4), 256, 0, stream>>>(logits, kv_cache, kv_new, top, btab, ctx, ohc);
  k_ov<<<dim3(128, 8), 32, 0, stream>>>(ohc, wkv_b_w, wkv_b_s, o2);
  k_quant_rows<<<dim3(4, 128), 128, 0, stream>>>(o2, o_f8, o_sc, 16384);
  k_gemm_fp8<<<448, 32, 0, stream>>>(o_f8, o_sc, wo_w, wo_s, (float*)d_out, 7168, 16384, 7168);
}
